// SearchTransfer_51470888075879
// MI455X (gfx1250) — matrix operand order repaired and checked by reading, not compiled, not run on hardware
//
#include <hip/hip_runtime.h>
#include <hip/hip_bf16.h>

// ---------- problem constants ----------
constexpr int kB    = 4;
constexpr int kC3   = 256;
constexpr int kH    = 48;
constexpr int kW    = 48;
constexpr int kL    = kH * kW;       // 2304 spatial positions
constexpr int kD    = kC3 * 9;       // 2304 patch dim (K)
constexpr int kNSEG = 9;             // l-segments for argmax parallelism
constexpr int kSEGL = kL / kNSEG;    // 256 l per segment
constexpr int kLT   = 64;            // l rows per block iteration (4 WMMA M-tiles)
constexpr int kKC   = 128;           // K chunk staged in LDS
constexpr int kAS   = 136;           // LDS A row stride in ushort (272B: 16B-aligned, conflict-free)

// ---------- vector types ----------
typedef __attribute__((ext_vector_type(16))) __bf16          v16bf;
typedef __attribute__((ext_vector_type(8)))  float           v8f;
typedef __attribute__((ext_vector_type(8)))  unsigned short  usv8;
typedef __attribute__((ext_vector_type(4)))  unsigned short  usv4;
typedef __attribute__((ext_vector_type(4)))  unsigned int    v4u;
typedef __attribute__((ext_vector_type(8)))  int             v8i;
typedef __attribute__((ext_vector_type(4)))  int             v4i;

union BFrag {
    v16bf bf;
    usv8  h[2];
    usv4  q[4];
};

__device__ __forceinline__ unsigned short f2bf(float f) {
    union { float f; unsigned u; } v; v.f = f;
    unsigned r = v.u + 0x7FFFu + ((v.u >> 16) & 1u);   // round-to-nearest-even
    return (unsigned short)(r >> 16);
}

// ---------- Tensor Data Mover path ----------
// Both toolchains declare __builtin_amdgcn_tensor_load_to_lds but with
// different arity: ROCm 7.2 (clang-22) = 5 args, amdgpu-toolchain
// (clang-23, ships the TDM header) = 6 args. Fingerprint via the header.
#if __has_builtin(__builtin_amdgcn_tensor_load_to_lds)
#define ATHENA_HAVE_TDM 1
#if __has_include(<hip/amd_detail/amd_gfx1250_TDM.h>)
#define ATHENA_TDM_ARITY 6
#else
#define ATHENA_TDM_ARITY 5
#endif
#else
#define ATHENA_HAVE_TDM 0
#endif

#if ATHENA_HAVE_TDM
// DMA a 64-row x 128-col bf16 tile (row stride kD elements) from global into
// LDS at byte offset ldsOff, inserting 4 DWORDs of padding after every 64
// DWORDs (256B row) -> LDS row stride 272B == kAS ushorts.
__device__ __forceinline__ void tdm_load_a(const unsigned short* gsrc, unsigned ldsOff)
{
    const unsigned long long ga = (unsigned long long)(size_t)gsrc;
    v4u g0;
    g0.x = 1u;                                              // count=1, user descriptor
    g0.y = ldsOff;                                          // lds_addr (bytes)
    g0.z = (unsigned)(ga & 0xFFFFFFFFu);                    // global_addr[31:0]
    g0.w = (unsigned)((ga >> 32) & 0x01FFFFFFu) | (2u << 30); // global_addr[56:32] | type=2
    v8i g1;
    g1[0] = (int)((1u << 16)          // data_size = 2 bytes
                | (1u << 20)          // pad_enable
                | (5u << 22)          // pad_interval: every 64 DWORDs
                | (3u << 25));        // pad_amount: 4 DWORDs
    g1[1] = (int)(((unsigned)kD & 0xFFFFu) << 16);          // tensor_dim0[15:0]
    g1[2] = (int)(((unsigned)kD >> 16) | (((unsigned)kL & 0xFFFFu) << 16)); // dim0 hi | dim1 lo
    g1[3] = (int)(((unsigned)kL >> 16) | (128u << 16));     // dim1 hi | tile_dim0 = 128
    g1[4] = (int)(64u);                                     // tile_dim1 = 64, tile_dim2 = 0
    g1[5] = (int)kD;                                        // tensor_dim0_stride[31:0]
    g1[6] = 0;                                              // stride hi, dim1_stride lo
    g1[7] = 0;
    const v4i gz = {0, 0, 0, 0};                            // 2-D tile: groups 2/3 unused
#if ATHENA_TDM_ARITY == 6
    const v8i gz8 = {0, 0, 0, 0, 0, 0, 0, 0};
    __builtin_amdgcn_tensor_load_to_lds(g0, g1, gz, gz, gz8, 0);
#else
    __builtin_amdgcn_tensor_load_to_lds(g0, g1, gz, gz, 0);
#endif
}
#endif

// =====================================================================
// Kernel 1: extract 3x3 patch per position, L2-normalize over D, emit bf16
// grid = B*L blocks of 256 threads (thread == channel)
// dst layout: [b*L + pos][d]  with d = c*9 + ky*3 + kx  (unfold order)
// =====================================================================
__global__ __launch_bounds__(256) void prep_patches(
    const float* __restrict__ src, unsigned short* __restrict__ dst)
{
    const int pos = blockIdx.x;          // b*L + m
    const int m   = pos % kL;
    const int b   = pos / kL;
    const int y   = m / kW;
    const int x   = m % kW;
    const int c   = threadIdx.x;         // 0..255

    const float* sp = src + ((size_t)(b * kC3 + c)) * (kH * kW);
    float vals[9];
    int j = 0;
    #pragma unroll
    for (int ky = 0; ky < 3; ++ky) {
        const int ry = y + ky - 1;
        #pragma unroll
        for (int kx = 0; kx < 3; ++kx) {
            const int rx = x + kx - 1;
            const bool ok = ((unsigned)ry < (unsigned)kH) && ((unsigned)rx < (unsigned)kW);
            vals[j++] = ok ? sp[ry * kW + rx] : 0.0f;
        }
    }
    float ss = 0.0f;
    #pragma unroll
    for (j = 0; j < 9; ++j) ss += vals[j] * vals[j];

    // block reduction: wave32 shuffle then LDS across 8 waves
    #pragma unroll
    for (int off = 16; off > 0; off >>= 1) ss += __shfl_down(ss, off);
    __shared__ float wsum[8];
    const int lane = threadIdx.x & 31, wid = threadIdx.x >> 5;
    if (lane == 0) wsum[wid] = ss;
    __syncthreads();
    if (threadIdx.x == 0) {
        float tot = 0.0f;
        #pragma unroll
        for (int i = 0; i < 8; ++i) tot += wsum[i];
        wsum[0] = tot;
    }
    __syncthreads();
    const float inv = 1.0f / fmaxf(sqrtf(wsum[0]), 1e-12f);

    unsigned short* dp = dst + (size_t)pos * kD + c * 9;
    #pragma unroll
    for (j = 0; j < 9; ++j) dp[j] = f2bf(vals[j] * inv);
}

// =====================================================================
// Kernel 2: streaming correlation GEMM + per-column running max/argmax
//   R[l,m] = dot(rsn[l,:], lrn[m,:])   (K = 2304, bf16 -> f32 WMMA)
// grid: x = m-block (18 of 128 cols), y = l-segment (9), z = batch
// block: 256 threads = 8 waves. Each wave owns a 16-col N tile with 4
// accumulators covering a 64-row l block; the 64 x 128 A chunk is shared
// through double-buffered LDS. Where available the chunk is DMA'd by the
// Tensor Data Mover (tensor_load_to_lds, TENSORcnt) with hardware row
// padding; otherwise a manual global->VGPR->LDS pipeline is used.
// Either way the copy of chunk i+1 overlaps the 16 WMMAs of chunk i.
// =====================================================================
__global__ __launch_bounds__(256) void gemm_argmax(
    const unsigned short* __restrict__ rsn,   // [B][L][D] (A rows)
    const unsigned short* __restrict__ lrn,   // [B][L][D] (B cols, K-major)
    float* __restrict__ pmax, int* __restrict__ parg)
{
    const int b    = blockIdx.z;
    const int seg  = blockIdx.y;
    const int mBlk = blockIdx.x;
    const int tid  = threadIdx.x;
    const int lane = tid & 31;
    const int wave = tid >> 5;            // 0..7
    const int nCol = lane & 15;           // N index within wave tile
    const int half = lane >> 4;           // 0/1 lane half
    const int aRow = lane & 15;           // M row within a 16-row subtile

    __shared__ unsigned short As[2][kLT * kAS];   // 2 x 8704 ushort = 34 KB

    const int mBase = mBlk * 128 + wave * 16;
    const unsigned short* Brow  = lrn + ((size_t)b * kL + (mBase + nCol)) * kD;
    const unsigned short* Abase = rsn + (size_t)b * kL * kD;

#if ATHENA_HAVE_TDM
    const unsigned ldsA0 = (unsigned)(size_t)(void*)&As[0][0];
    const unsigned ldsA1 = (unsigned)(size_t)(void*)&As[1][0];
#else
    // cooperative A staging: 256 threads -> 64 rows x (4 x 32-col) chunks
    const int sRow = tid >> 2;            // 0..63
    const int sCol = (tid & 3) * 32;      // 0,32,64,96
    unsigned short* sPtr = &As[0][sRow * kAS + sCol];
    unsigned short* sPtrAlt = &As[1][sRow * kAS + sCol];
#endif

    float bestV = -3.0e38f;
    int   bestL = 0;

    constexpr int NCHUNK = kD / kKC;      // 18

    for (int lb = seg * kSEGL; lb < seg * kSEGL + kSEGL; lb += kLT) {
        v8f acc0 = {}, acc1 = {}, acc2 = {}, acc3 = {};

#if ATHENA_HAVE_TDM
        const unsigned short* Atile = Abase + (size_t)lb * kD;
        if (wave == 0) {
            tdm_load_a(Atile, ldsA0);
            __builtin_amdgcn_s_wait_tensorcnt((short)0);
        }
        __syncthreads();
#else
        const unsigned short* Ath = Abase + (size_t)(lb + sRow) * kD + sCol;
        {
            usv8 s0 = *(const usv8*)(Ath + 0);
            usv8 s1 = *(const usv8*)(Ath + 8);
            usv8 s2 = *(const usv8*)(Ath + 16);
            usv8 s3 = *(const usv8*)(Ath + 24);
            *(usv8*)(sPtr + 0)  = s0;
            *(usv8*)(sPtr + 8)  = s1;
            *(usv8*)(sPtr + 16) = s2;
            *(usv8*)(sPtr + 24) = s3;
        }
        __syncthreads();
#endif

        for (int ch = 0; ch < NCHUNK; ++ch) {
            const int cur = ch & 1;
            const bool hasNext = (ch + 1) < NCHUNK;
            const int kGlob = ch * kKC;

#if ATHENA_HAVE_TDM
            // DMA next chunk into the other buffer; overlaps the WMMAs below
            if (hasNext) {
                if (wave == 0)
                    tdm_load_a(Atile + (ch + 1) * kKC, cur ? ldsA0 : ldsA1);
                __builtin_prefetch(Brow + kGlob + kKC, 0, 0);
            }
#else
            // issue next chunk's global loads early (overlap with WMMAs)
            usv8 s0, s1, s2, s3;
            if (hasNext) {
                const unsigned short* An = Ath + (ch + 1) * kKC;
                s0 = *(const usv8*)(An + 0);
                s1 = *(const usv8*)(An + 8);
                s2 = *(const usv8*)(An + 16);
                s3 = *(const usv8*)(An + 24);
                __builtin_prefetch(Brow + kGlob + kKC, 0, 0);
            }
#endif

            const unsigned short* buf = &As[cur][0];
            #pragma unroll
            for (int kk = 0; kk < kKC; kk += 32) {
                // B fragment (32x16): lane half h holds K 8h..8h+7 and 16+8h..16+8h+7
                BFrag bb;
                const unsigned short* br = Brow + kGlob + kk + half * 8;
                bb.h[0] = *(const usv8*)(br);
                bb.h[1] = *(const usv8*)(br + 16);

                // 4 M-subtiles reuse the B fragment
                {
                    BFrag a;
                    const unsigned short* ar = buf + (0 * 16 + aRow) * kAS + kk + half * 8;
                    a.h[0] = *(const usv8*)(ar);
                    a.h[1] = *(const usv8*)(ar + 16);
                    acc0 = __builtin_amdgcn_wmma_f32_16x16x32_bf16(
                        false, a.bf, false, bb.bf, (short)0, acc0, false, false);
                }
                {
                    BFrag a;
                    const unsigned short* ar = buf + (1 * 16 + aRow) * kAS + kk + half * 8;
                    a.h[0] = *(const usv8*)(ar);
                    a.h[1] = *(const usv8*)(ar + 16);
                    acc1 = __builtin_amdgcn_wmma_f32_16x16x32_bf16(
                        false, a.bf, false, bb.bf, (short)0, acc1, false, false);
                }
                {
                    BFrag a;
                    const unsigned short* ar = buf + (2 * 16 + aRow) * kAS + kk + half * 8;
                    a.h[0] = *(const usv8*)(ar);
                    a.h[1] = *(const usv8*)(ar + 16);
                    acc2 = __builtin_amdgcn_wmma_f32_16x16x32_bf16(
                        false, a.bf, false, bb.bf, (short)0, acc2, false, false);
                }
                {
                    BFrag a;
                    const unsigned short* ar = buf + (3 * 16 + aRow) * kAS + kk + half * 8;
                    a.h[0] = *(const usv8*)(ar);
                    a.h[1] = *(const usv8*)(ar + 16);
                    acc3 = __builtin_amdgcn_wmma_f32_16x16x32_bf16(
                        false, a.bf, false, bb.bf, (short)0, acc3, false, false);
                }
            }

#if ATHENA_HAVE_TDM
            if (hasNext && wave == 0)
                __builtin_amdgcn_s_wait_tensorcnt((short)0);
#else
            // park next chunk into the other buffer
            if (hasNext) {
                unsigned short* dst = cur ? sPtr : sPtrAlt;
                *(usv8*)(dst + 0)  = s0;
                *(usv8*)(dst + 8)  = s1;
                *(usv8*)(dst + 16) = s2;
                *(usv8*)(dst + 24) = s3;
            }
#endif
            __syncthreads();
        }

        // fold the 64x16 result block into per-lane running max
        #pragma unroll
        for (int v = 0; v < 8; ++v) {
            const float r0 = acc0[v]; const int l0 = lb +  0 + v + 8 * half;
            if (r0 > bestV || (r0 == bestV && l0 < bestL)) { bestV = r0; bestL = l0; }
            const float r1 = acc1[v]; const int l1 = lb + 16 + v + 8 * half;
            if (r1 > bestV || (r1 == bestV && l1 < bestL)) { bestV = r1; bestL = l1; }
            const float r2 = acc2[v]; const int l2 = lb + 32 + v + 8 * half;
            if (r2 > bestV || (r2 == bestV && l2 < bestL)) { bestV = r2; bestL = l2; }
            const float r3 = acc3[v]; const int l3 = lb + 48 + v + 8 * half;
            if (r3 > bestV || (r3 == bestV && l3 < bestL)) { bestV = r3; bestL = l3; }
        }
    }

    // combine the two lane halves that share the same N column
    const float ov = __shfl_xor(bestV, 16);
    const int   ol = __shfl_xor(bestL, 16);
    if (ov > bestV || (ov == bestV && ol < bestL)) { bestV = ov; bestL = ol; }

    if (half == 0) {
        const int m = mBase + nCol;
        const size_t off = ((size_t)(b * kNSEG + seg)) * kL + m;
        pmax[off] = bestV;
        parg[off] = bestL;
    }
}

// =====================================================================
// Kernel 3: reduce partial (max,arg) over segments -> S and final arg
// =====================================================================
__global__ void reduce_arg(const float* __restrict__ pmax,
                           const int* __restrict__ parg,
                           float* __restrict__ S, int* __restrict__ argF)
{
    const int i = blockIdx.x * blockDim.x + threadIdx.x;   // b*L + m
    if (i >= kB * kL) return;
    const int b = i / kL, m = i - b * kL;
    float bv = -3.0e38f; int bl = 0;
    #pragma unroll
    for (int s = 0; s < kNSEG; ++s) {
        const size_t off = ((size_t)(b * kNSEG + s)) * kL + m;
        const float v = pmax[off]; const int a = parg[off];
        if (v > bv || (v == bv && a < bl)) { bv = v; bl = a; }
    }
    S[i]    = bv;     // S is (B,1,48,48) flat
    argF[i] = bl;
}

// =====================================================================
// Kernels 4-6: unfold->gather->fold collapsed to direct 9-tap gather.
// scale s in {1,2,4}: out[c,Y,X] = (1/9) * sum over 3x3 covering patches of
//   ref[c, s*argY + (Y - s*yy), s*argX + (X - s*xx)]
// One block per (b,Y,X); threads = channels; 9 lookups cached in LDS.
// =====================================================================
template <int SCALE, int CH, int OHW>
__global__ void fold_gather(const float* __restrict__ ref,
                            const int* __restrict__ argF,
                            float* __restrict__ out)
{
    const int X = blockIdx.x, Y = blockIdx.y, b = blockIdx.z;
    const int c = threadIdx.x;

    __shared__ int s_ry[9], s_rx[9], s_ok[9];
    if (c < 9) {
        const int oy = c / 3 - 1, ox = c % 3 - 1;
        const int yy = Y / SCALE + oy, xx = X / SCALE + ox;
        int ry = 0, rx = 0, ok = 0;
        if ((unsigned)yy < (unsigned)kH && (unsigned)xx < (unsigned)kW) {
            const int p  = argF[b * kL + yy * kW + xx];
            ry = SCALE * (p / kW) + (Y - SCALE * yy);
            rx = SCALE * (p % kW) + (X - SCALE * xx);
            ok = ((unsigned)ry < (unsigned)OHW) && ((unsigned)rx < (unsigned)OHW);
        }
        s_ry[c] = ry; s_rx[c] = rx; s_ok[c] = ok;
    }
    __syncthreads();

    const float* rp = ref + ((size_t)(b * CH + c)) * (OHW * OHW);
    float s = 0.0f;
    #pragma unroll
    for (int j = 0; j < 9; ++j)
        if (s_ok[j]) s += rp[s_ry[j] * OHW + s_rx[j]];

    out[(((size_t)(b * CH + c)) * OHW + Y) * OHW + X] = s * (1.0f / 9.0f);
}

// =====================================================================
// host launcher
// =====================================================================
extern "C" void kernel_launch(void* const* d_in, const int* in_sizes, int n_in,
                              void* d_out, int out_size, void* d_ws, size_t ws_size,
                              hipStream_t stream)
{
    const float* lrsr = (const float*)d_in[0];   // (4,256,48,48)
    const float* refsr = (const float*)d_in[1];  // (4,256,48,48)
    const float* ref1 = (const float*)d_in[2];   // (4,64,192,192)
    const float* ref2 = (const float*)d_in[3];   // (4,128,96,96)
    const float* ref3 = (const float*)d_in[4];   // (4,256,48,48)
    float* out = (float*)d_out;

    // ---- workspace carving ----
    char* ws = (char*)d_ws;
    const size_t szMat = (size_t)kB * kL * kD * sizeof(unsigned short); // 42.5 MB
    size_t off = 0;
    unsigned short* rsn = (unsigned short*)(ws + off); off += szMat;
    unsigned short* lrn = (unsigned short*)(ws + off); off += szMat;
    float* pmax = (float*)(ws + off); off += (size_t)kB * kNSEG * kL * sizeof(float);
    int*   parg = (int*)(ws + off);   off += (size_t)kB * kNSEG * kL * sizeof(int);
    int*   argF = (int*)(ws + off);   off += (size_t)kB * kL * sizeof(int);
    (void)off; (void)ws_size; (void)in_sizes; (void)n_in; (void)out_size;

    // ---- output regions (return order: S, T_lv3, T_lv2, T_lv1) ----
    float* S  = out;
    float* T3 = out + (size_t)kB * kL;                               // 9216
    float* T2 = T3 + (size_t)kB * kC3 * kH * kW;                     // +2359296
    float* T1 = T2 + (size_t)kB * 128 * 96 * 96;                     // +4718592

    // 1) normalized bf16 patch matrices
    prep_patches<<<kB * kL, 256, 0, stream>>>(refsr, rsn);
    prep_patches<<<kB * kL, 256, 0, stream>>>(lrsr, lrn);

    // 2) correlation GEMM with streaming argmax
    gemm_argmax<<<dim3(kL / 128, kNSEG, kB), 256, 0, stream>>>(rsn, lrn, pmax, parg);

    // 3) segment reduction -> S, argF
    reduce_arg<<<(kB * kL + 255) / 256, 256, 0, stream>>>(pmax, parg, S, argF);

    // 4) transfer (gather + overlap-add fold) at each level
    fold_gather<1, 256, 48 ><<<dim3(48, 48, kB), 256, 0, stream>>>(ref3, argF, T3);
    fold_gather<2, 128, 96 ><<<dim3(96, 96, kB), 128, 0, stream>>>(ref2, argF, T2);
    fold_gather<4, 64, 192><<<dim3(192, 192, kB), 64, 0, stream>>>(ref1, argF, T1);
}
